// EnhancedVLAD_64269890617681
// MI455X (gfx1250) — matrix operand order repaired and checked by reading, not compiled, not run on hardware
//
#include <hip/hip_runtime.h>
#include <cmath>

typedef __attribute__((ext_vector_type(16))) _Float16 v16h;
typedef __attribute__((ext_vector_type(8)))  float    v8f;

#define BB   32
#define CC   512
#define NN   4096
#define TT   72     // total clusters
#define TP   80     // padded to 5 WMMA tiles
#define KCL  64     // real clusters
#define HH   18     // attention hidden
#define SS   8      // N splits (blocks per batch)
#define NT   128    // n-columns per LDS tile
#define TIT  4      // tiles per block: SS*TIT*NT == NN
#define NWV  16     // waves per block (512 threads, wave32)
#define XH2S 520    // padded halves stride for n-major x tile (16B aligned rows)
#define EPSN 1e-12f

union H16 { v16h v; _Float16 h[16]; };

// LDS carve (bytes):
//   xh1 [512][128] f16 (c-major, GEMM2 B)        @0       131072
//   xh2 [128][520] f16 (n-major padded, GEMM1 A) @131072  133120
//   lgT [128][80]  f32 (logits^T -> assign)      @264192   40960  (sumsq scratch aliases)
//   asg [80][128]  f16 (assign * 1/||x||)        @305152   20480
//   sc  [128] f32                                @325632     512
//   cb  [80]  f32                                @326144     320
#define SM_XH1  0
#define SM_XH2  131072
#define SM_LGT  264192
#define SM_ASG  305152
#define SM_SC   325632
#define SM_CB   326144
#define SM_K1   326464

// ---- kernel 0: one-time conv_w f32 -> f16 (L2-resident operand for GEMM1 B) ----
__global__ __launch_bounds__(256) void conv_cvt_kernel(
    const float* __restrict__ conv_w, _Float16* __restrict__ cwh)
{
    const int idx = blockIdx.x * 256 + threadIdx.x;
    if (idx < TP * CC)
        cwh[idx] = (idx < TT * CC) ? (_Float16)conv_w[idx] : (_Float16)0.0f;
}

__global__ __launch_bounds__(512) void vlad_main_kernel(
    const float* __restrict__ x, const _Float16* __restrict__ cwh,
    const float* __restrict__ conv_b, float* __restrict__ pAgg,
    float* __restrict__ pMass)
{
    extern __shared__ char smem[];
    _Float16* xh1 = (_Float16*)(smem + SM_XH1);
    _Float16* xh2 = (_Float16*)(smem + SM_XH2);
    float*    lgT = (float*)   (smem + SM_LGT);
    float*    scr = (float*)   (smem + SM_LGT);   // alias: dead before GEMM1 writes
    _Float16* asg = (_Float16*)(smem + SM_ASG);
    float*    sc  = (float*)   (smem + SM_SC);
    float*    cb  = (float*)   (smem + SM_CB);

    const int tid  = threadIdx.x;
    const int w    = tid >> 5;       // wave 0..15
    const int lane = tid & 31;
    const int l15  = lane & 15;
    const int hi   = lane >> 4;
    const int b    = blockIdx.x / SS;
    const int s    = blockIdx.x % SS;
    const int col  = tid & (NT - 1);
    const int crow = tid >> 7;       // 0..3

    for (int idx = tid; idx < TP * NT; idx += 512)
        asg[idx] = (_Float16)0.0f;                 // rows 72..79 stay zero
    if (tid < TT) cb[tid] = conv_b[tid];

    v8f acc2[10];                                  // persistent GEMM2 accumulators
    #pragma unroll
    for (int i = 0; i < 10; ++i) acc2[i] = (v8f){0,0,0,0,0,0,0,0};
    float massReg[5] = {0.f, 0.f, 0.f, 0.f, 0.f};
    __syncthreads();

    for (int t = 0; t < TIT; ++t) {
        const int n0 = (s * TIT + t) * NT;

        // ---- phase A: stream x tile once from HBM; write both LDS layouts ----
        const float* xb = x + (size_t)b * CC * NN + n0 + col;
        if (t + 1 < TIT) __builtin_prefetch(xb + NT, 0, 0);
        float ss = 0.f;
        #pragma unroll 4
        for (int c = crow; c < CC; c += 4) {
            float v = xb[(size_t)c * NN];
            xh1[c * NT + col]  = (_Float16)v;      // c-major (GEMM2 B)
            xh2[col * XH2S + c] = (_Float16)v;     // n-major (GEMM1 A)
            ss += v * v;
        }
        scr[tid] = ss;
        __syncthreads();
        if (tid < NT) {
            float tot = scr[tid] + scr[tid + 128] + scr[tid + 256] + scr[tid + 384];
            sc[tid] = 1.0f / fmaxf(sqrtf(tot), EPSN);
        }
        __syncthreads();

        // ---- GEMM1: logits^T[128][80] = x_tile^T[128][512] * conv_w^T[512][80] ----
        // A = xh2 (M=n, K=c), B = cwh rows (K=c contiguous, N=cluster) from L2.
        for (int ti = w; ti < 40; ti += NWV) {
            const int nt = ti & 7, kt = ti >> 3;
            const int arow = nt * 16 + l15;        // n within tile
            const int kcol = kt * 16 + l15;        // cluster
            v8f acc = (v8f){0,0,0,0,0,0,0,0};
            #pragma unroll 4
            for (int kk = 0; kk < 16; ++kk) {
                H16 au;
                const _Float16* ap = &xh2[arow * XH2S + kk * 32 + hi * 16];
                #pragma unroll
                for (int j = 0; j < 8; ++j) { au.h[j] = ap[j]; au.h[8 + j] = ap[8 + j]; }
                const v16h bv = *(const v16h*)&cwh[kcol * CC + kk * 32 + hi * 16];
                acc = __builtin_amdgcn_wmma_f32_16x16x32_f16(false, au.v, false, bv,
                                                             (short)0, acc, false, false);
            }
            #pragma unroll
            for (int v = 0; v < 8; ++v)
                lgT[(nt * 16 + hi * 8 + v) * TP + kt * 16 + l15] = acc[v];
        }
        __syncthreads();

        // ---- softmax over 72 clusters per column (row of lgT); fold scale+bias ----
        if (tid < NT) {
            const float scv = sc[tid];
            float* rowp = &lgT[tid * TP];
            float mx = -3.0e38f;
            #pragma unroll 4
            for (int k = 0; k < TT; ++k)
                mx = fmaxf(mx, rowp[k] * scv + cb[k]);
            float sum = 0.f;
            #pragma unroll 4
            for (int k = 0; k < TT; ++k) {
                float e = __expf(rowp[k] * scv + cb[k] - mx);
                rowp[k] = e;
                sum += e;
            }
            const float inv = 1.0f / sum;
            #pragma unroll 4
            for (int k = 0; k < TT; ++k) {
                float a = rowp[k] * inv;
                rowp[k] = a;                          // f32 assign (for mass)
                asg[k * NT + tid] = (_Float16)(a * scv);  // fold 1/||x|| into A-side
            }
        }
        __syncthreads();

        // ---- mass[k] += sum_n assign (wave shuffle reduce, deterministic) ----
        #pragma unroll
        for (int i = 0; i < 5; ++i) {
            const int k = w + NWV * i;
            if (k < TT) {
                float p = 0.f;
                for (int n = lane; n < NT; n += 32) p += lgT[n * TP + k];
                #pragma unroll
                for (int off = 16; off; off >>= 1) p += __shfl_xor(p, off, 32);
                massReg[i] += p;
            }
        }

        // ---- GEMM2: agg[80][512] += asg[80][128] * x_tile^T[128][512] ----
        #pragma unroll
        for (int i = 0; i < 10; ++i) {
            const int ti = w + NWV * i;
            const int mt = ti >> 5, ct = ti & 31;
            const int arow = mt * 16 + l15;
            const int ccol = ct * 16 + l15;
            v8f acc = acc2[i];
            #pragma unroll
            for (int kk = 0; kk < 4; ++kk) {
                H16 au;
                const _Float16* ap = &asg[arow * NT + kk * 32 + hi * 16];
                #pragma unroll
                for (int j = 0; j < 8; ++j) { au.h[j] = ap[j]; au.h[8 + j] = ap[8 + j]; }
                const v16h bv = *(const v16h*)&xh1[ccol * NT + kk * 32 + hi * 16];
                acc = __builtin_amdgcn_wmma_f32_16x16x32_f16(false, au.v, false, bv,
                                                             (short)0, acc, false, false);
            }
            acc2[i] = acc;
        }
        __syncthreads();   // protect xh1/xh2/lgT/asg before next tile
    }

    // ---- write partials ----
    float* base = pAgg + (size_t)(b * SS + s) * TP * CC;
    #pragma unroll
    for (int i = 0; i < 10; ++i) {
        const int ti = w + NWV * i;
        const int mt = ti >> 5, ct = ti & 31;
        const int ccol = ct * 16 + l15;
        #pragma unroll
        for (int v = 0; v < 8; ++v)
            base[(mt * 16 + hi * 8 + v) * CC + ccol] = acc2[i][v];
    }
    #pragma unroll
    for (int i = 0; i < 5; ++i) {
        const int k = w + NWV * i;
        if (k < TT && lane == 0) pMass[(b * SS + s) * TP + k] = massReg[i];
    }
}

// ---- kernel 2: reduce partials + centroids + ghosts + attention MLP + norms ----
#define SM_K2 ((36864 + 72 + 18 + 72 + 8 + 72 + 8 + 72 + 1) * 4)

__global__ __launch_bounds__(256) void vlad_finish_kernel(
    const float* __restrict__ pAgg, const float* __restrict__ pMass,
    const float* __restrict__ centroids, const float* __restrict__ gweights,
    const float* __restrict__ w1, const float* __restrict__ b1,
    const float* __restrict__ w2, const float* __restrict__ b2,
    float* __restrict__ out)
{
    extern __shared__ char smem2[];
    float* vlad    = (float*)smem2;            // [72][512]
    float* scoresL = vlad + TT * CC;
    float* hA      = scoresL + TT;
    float* attnL   = hA + HH;
    float* gwL     = attnL + TT;
    float* massL   = gwL + 8;
    float* gpart   = massL + TT;
    float* rnL     = gpart + 8;
    float* gInvL   = rnL + TT;

    const int bI   = blockIdx.x;
    const int tid  = threadIdx.x;
    const int w    = tid >> 5;
    const int lane = tid & 31;

    if (tid < TT) {
        float m = 0.f;
        for (int s = 0; s < SS; ++s) m += pMass[(bI * SS + s) * TP + tid];
        massL[tid] = m;
    }
    if (tid == 0) {   // ghost softmax (8 values)
        float mx = -3.0e38f;
        for (int i = 0; i < 8; ++i) mx = fmaxf(mx, gweights[i]);
        float sm = 0.f;
        for (int i = 0; i < 8; ++i) { gwL[i] = __expf(gweights[i] - mx); sm += gwL[i]; }
        for (int i = 0; i < 8; ++i) gwL[i] /= sm;
    }
    __syncthreads();

    for (int idx = tid; idx < TT * CC; idx += 256) {
        const int k = idx >> 9;
        float a = 0.f;
        for (int s = 0; s < SS; ++s) a += pAgg[(size_t)(bI * SS + s) * TP * CC + idx];
        float v = a - centroids[idx] * massL[k];
        if (k >= KCL) v *= gwL[k - KCL];
        vlad[idx] = v;
    }
    __syncthreads();

    for (int k = w; k < TT; k += 8) {      // scores = mean over C
        float p = 0.f;
        for (int c = lane; c < CC; c += 32) p += vlad[(k << 9) + c];
        #pragma unroll
        for (int off = 16; off; off >>= 1) p += __shfl_xor(p, off, 32);
        if (lane == 0) scoresL[k] = p * (1.0f / (float)CC);
    }
    __syncthreads();
    if (tid < HH) {
        float hv = b1[tid];
        for (int k = 0; k < TT; ++k) hv += scoresL[k] * w1[tid * TT + k];
        hA[tid] = fmaxf(hv, 0.f);
    }
    __syncthreads();
    if (tid < TT) {
        float o = b2[tid];
        for (int j = 0; j < HH; ++j) o += hA[j] * w2[tid * HH + j];
        attnL[tid] = o;
    }
    __syncthreads();
    if (tid == 0) {   // cluster softmax
        float mx = -3.0e38f;
        for (int k = 0; k < TT; ++k) mx = fmaxf(mx, attnL[k]);
        float sm = 0.f;
        for (int k = 0; k < TT; ++k) { attnL[k] = __expf(attnL[k] - mx); sm += attnL[k]; }
        const float inv = 1.0f / sm;
        for (int k = 0; k < TT; ++k) attnL[k] *= inv;
    }
    __syncthreads();

    float gp = 0.f;
    for (int k = w; k < KCL; k += 8) {     // per-cluster norms + global sumsq
        const float at = attnL[k];
        float p = 0.f;
        for (int c = lane; c < CC; c += 32) {
            float v = vlad[(k << 9) + c] * at;
            p += v * v;
        }
        #pragma unroll
        for (int off = 16; off; off >>= 1) p += __shfl_xor(p, off, 32);
        const float rn = 1.0f / fmaxf(sqrtf(p), EPSN);
        if (lane == 0) rnL[k] = rn;
        gp += p * rn * rn;
    }
    if (lane == 0) gpart[w] = gp;
    __syncthreads();
    if (tid == 0) {
        float g = 0.f;
        for (int i = 0; i < 8; ++i) g += gpart[i];
        gInvL[0] = 1.0f / fmaxf(sqrtf(g), EPSN);
    }
    __syncthreads();

    const float gi = gInvL[0];
    for (int idx = tid; idx < KCL * CC; idx += 256) {
        const int k = idx >> 9;
        out[(size_t)bI * (KCL * CC) + idx] = vlad[idx] * attnL[k] * rnL[k] * gi;
    }
}

extern "C" void kernel_launch(void* const* d_in, const int* in_sizes, int n_in,
                              void* d_out, int out_size, void* d_ws, size_t ws_size,
                              hipStream_t stream) {
    const float* x         = (const float*)d_in[0];
    const float* centroids = (const float*)d_in[1];
    const float* conv_w    = (const float*)d_in[2];
    const float* conv_b    = (const float*)d_in[3];
    const float* gweights  = (const float*)d_in[4];
    const float* w1        = (const float*)d_in[5];
    const float* b1        = (const float*)d_in[6];
    const float* w2        = (const float*)d_in[7];
    const float* b2        = (const float*)d_in[8];
    float* out = (float*)d_out;

    float*     pAgg  = (float*)d_ws;                        // [B][S][80][512] f32
    float*     pMass = pAgg + (size_t)BB * SS * TP * CC;    // [B][S][80] f32
    _Float16*  cwh   = (_Float16*)(pMass + (size_t)BB * SS * TP);  // [80][512] f16

    conv_cvt_kernel<<<dim3((TP * CC + 255) / 256), dim3(256), 0, stream>>>(conv_w, cwh);
    vlad_main_kernel<<<dim3(BB * SS), dim3(512), SM_K1, stream>>>(
        x, cwh, conv_b, pAgg, pMass);
    vlad_finish_kernel<<<dim3(BB), dim3(256), SM_K2, stream>>>(
        pAgg, pMass, centroids, gweights, w1, b1, w2, b2, out);
}
